// MUSIC_transformer_model_35450660061590
// MI455X (gfx1250) — matrix operand order repaired and checked by reading, not compiled, not run on hardware
//
#include <hip/hip_runtime.h>

typedef unsigned short ushort_t;
typedef __attribute__((ext_vector_type(16))) __bf16 v16bf;
typedef __attribute__((ext_vector_type(8)))  float  v8f;

struct alignas(16) U4 { unsigned int a, b, c, d; };
struct alignas(8)  U2 { unsigned int a, b; };

union Frag {
  v16bf    v;
  ushort_t u[16];
  U4       q[2];
  U2       h2[4];
};

static constexpr int B_ = 8, T_ = 1024, C_ = 384, H_ = 16, D_ = 24, F_ = 1536, L_ = 8;
static constexpr int M_ = B_ * T_;   // 8192 rows of activations

__device__ __forceinline__ ushort_t f32_to_bf16(float f) {
  union { float f; unsigned int u; } cv; cv.f = f;
  unsigned int u = cv.u;
  u += 0x7fffu + ((u >> 16) & 1u);   // round-to-nearest-even
  return (ushort_t)(u >> 16);
}

__device__ __forceinline__ v8f v8f_zero() {
  v8f z;
#pragma unroll
  for (int i = 0; i < 8; ++i) z[i] = 0.0f;
  return z;
}

// ---------------------------------------------------------------------------
// x = idx + pos_emb (broadcast over batch), fp32
// ---------------------------------------------------------------------------
__global__ void add_pos_kernel(const float* __restrict__ idx,
                               const float* __restrict__ pos,
                               float* __restrict__ x) {
  int i = blockIdx.x * 256 + threadIdx.x;      // over B*T*C (exact grid)
  const int per = T_ * C_;
  int bi = i / per;
  int r  = i - bi * per;
  x[i] = idx[i] + pos[r];
}

// ---------------------------------------------------------------------------
// LayerNorm -> bf16 (one wave per row, C=384 -> 12 elems/lane)
// ---------------------------------------------------------------------------
__global__ __launch_bounds__(256) void ln_to_bf16_kernel(
    const float* __restrict__ x, const float* __restrict__ g,
    const float* __restrict__ b, ushort_t* __restrict__ out) {
  int row  = blockIdx.x * 8 + (threadIdx.x >> 5);
  int lane = threadIdx.x & 31;
  const float* xr = x + (size_t)row * C_;
  float vals[12];
  float s = 0.f, s2 = 0.f;
#pragma unroll
  for (int i = 0; i < 12; ++i) {
    float v = xr[lane + i * 32];
    vals[i] = v; s += v; s2 += v * v;
  }
#pragma unroll
  for (int m = 16; m >= 1; m >>= 1) { s += __shfl_xor(s, m); s2 += __shfl_xor(s2, m); }
  float mean = s * (1.0f / C_);
  float var  = s2 * (1.0f / C_) - mean * mean;
  float rstd = rsqrtf(var + 1e-5f);
  ushort_t* orow = out + (size_t)row * C_;
#pragma unroll
  for (int i = 0; i < 12; ++i) {
    int c = lane + i * 32;
    orow[c] = f32_to_bf16((vals[i] - mean) * rstd * g[c] + b[c]);
  }
}

// Final LN + ReLU -> fp32 d_out
__global__ __launch_bounds__(256) void lnf_relu_kernel(
    const float* __restrict__ x, const float* __restrict__ g,
    const float* __restrict__ b, float* __restrict__ out) {
  int row  = blockIdx.x * 8 + (threadIdx.x >> 5);
  int lane = threadIdx.x & 31;
  const float* xr = x + (size_t)row * C_;
  float vals[12];
  float s = 0.f, s2 = 0.f;
#pragma unroll
  for (int i = 0; i < 12; ++i) {
    float v = xr[lane + i * 32];
    vals[i] = v; s += v; s2 += v * v;
  }
#pragma unroll
  for (int m = 16; m >= 1; m >>= 1) { s += __shfl_xor(s, m); s2 += __shfl_xor(s2, m); }
  float mean = s * (1.0f / C_);
  float var  = s2 * (1.0f / C_) - mean * mean;
  float rstd = rsqrtf(var + 1e-5f);
  float* orow = out + (size_t)row * C_;
#pragma unroll
  for (int i = 0; i < 12; ++i) {
    int c = lane + i * 32;
    orow[c] = fmaxf((vals[i] - mean) * rstd * g[c] + b[c], 0.0f);
  }
}

// ---------------------------------------------------------------------------
// Weight packing: Wq/Wk/Wv (H,C,D) fp32 -> [C, H*D] bf16
// ---------------------------------------------------------------------------
__global__ void pack_qkv_kernel(const float* __restrict__ W, ushort_t* __restrict__ out) {
  int i = blockIdx.x * 256 + threadIdx.x;
  if (i >= C_ * H_ * D_) return;
  int c  = i / (H_ * D_);
  int hd = i - c * (H_ * D_);
  int h  = hd / D_;
  int d  = hd - h * D_;
  out[i] = f32_to_bf16(W[(size_t)h * C_ * D_ + (size_t)c * D_ + d]);
}

__global__ void pack_mat_kernel(const float* __restrict__ W, ushort_t* __restrict__ out, int n) {
  int i = blockIdx.x * 256 + threadIdx.x;
  if (i < n) out[i] = f32_to_bf16(W[i]);
}

// ---------------------------------------------------------------------------
// Generic bf16 WMMA GEMM: out[M,N] = A[M,K] @ B[K,N] (+bias)(+fp32 residual)(relu)
// 256 threads = 8 waves; macro tile 128x64; each wave: 2x2 WMMA tiles of 16x16.
// Double-buffered LDS; A-tile staged with CDNA5 async global->LDS DMA
// (global_load_async_to_lds_b128, ASYNCcnt), B-tile transposed through VGPRs.
// ---------------------------------------------------------------------------
template <bool BIAS, bool RES, bool RELU, bool OUTBF>
__global__ __launch_bounds__(256) void gemm_bf16_kernel(
    const ushort_t* __restrict__ A, const ushort_t* __restrict__ Bw,
    const float* __restrict__ bias, const float* __restrict__ resid,
    void* __restrict__ outp, int M, int N, int K) {
  constexpr int BM = 128, BN = 64, BK = 32;
  __shared__ alignas(16) ushort_t As[2][BM][BK + 8];   // [buf][m][k]
  __shared__ alignas(16) ushort_t Bt[2][BN][BK + 8];   // [buf][n][k] (transposed)

  const int tid   = threadIdx.x;
  const int wave  = tid >> 5;
  const int lane  = tid & 31;
  const int half  = lane >> 4;
  const int l16   = lane & 15;
  const int bm    = blockIdx.y * BM;
  const int bn    = blockIdx.x * BN;
  const int waveM = wave >> 1;   // 0..3
  const int waveN = wave & 1;    // 0..1

  v8f acc[2][2];
  acc[0][0] = v8f_zero(); acc[0][1] = v8f_zero();
  acc[1][0] = v8f_zero(); acc[1][1] = v8f_zero();

  // Stage one 128x32 A tile (async DMA to LDS) + 32x64 B tile (reg transpose).
  auto stage = [&](int k0, int buf) {
    // A tile: 512 chunks of 8 bf16 (16B); async global->LDS, no VGPR round-trip.
    // LDS aperture maps addr[31:0] directly to the LDS byte offset.
#pragma unroll
    for (int it = 0; it < 2; ++it) {
      int ch = tid + it * 256;
      int r  = ch >> 2;
      int cb = (ch & 3) * 8;
      unsigned lds_off = (unsigned)(size_t)&As[buf][r][cb];
      const ushort_t* gp = A + (size_t)(bm + r) * K + k0 + cb;
      asm volatile("global_load_async_to_lds_b128 %0, %1, off"
                   :: "v"(lds_off), "v"(gp) : "memory");
    }
    // B tile 32x64 -> transposed into Bt[buf][n][k]: 256 chunks of 8
    {
      int kk = tid >> 3;
      int nb = (tid & 7) * 8;
      U4 vq = *(const U4*)(Bw + (size_t)(k0 + kk) * N + bn + nb);
      ushort_t tmp[8];
      *(U4*)tmp = vq;
#pragma unroll
      for (int j = 0; j < 8; ++j) Bt[buf][nb + j][kk] = tmp[j];
    }
  };

  const int NT = K / BK;
  stage(0, 0);

  for (int kt = 0; kt < NT; ++kt) {
    const int cur = kt & 1;
    // Wave's async LDS writes complete, then publish all waves' stores.
    asm volatile("s_wait_asynccnt 0x0" ::: "memory");
    __syncthreads();
    if (kt + 1 < NT) stage((kt + 1) * BK, cur ^ 1);   // prefetch into other buffer

    Frag af[2], bf[2];
#pragma unroll
    for (int mi = 0; mi < 2; ++mi) {
      int r  = waveM * 32 + mi * 16 + l16;
      int kb = half * 8;                       // ISA A layout: {0..7,16..23}/{8..15,24..31}
      af[mi].q[0] = *(const U4*)&As[cur][r][kb];
      af[mi].q[1] = *(const U4*)&As[cur][r][kb + 16];
    }
#pragma unroll
    for (int ni = 0; ni < 2; ++ni) {
      int c  = waveN * 32 + ni * 16 + l16;
      int kf = half * 8;                       // B layout: col=lane%16, K {0..7,16..23}/{8..15,24..31}
      bf[ni].q[0] = *(const U4*)&Bt[cur][c][kf];
      bf[ni].q[1] = *(const U4*)&Bt[cur][c][kf + 16];
    }
#pragma unroll
    for (int mi = 0; mi < 2; ++mi)
#pragma unroll
      for (int ni = 0; ni < 2; ++ni)
        acc[mi][ni] = __builtin_amdgcn_wmma_f32_16x16x32_bf16(
            false, af[mi].v, false, bf[ni].v, (short)0, acc[mi][ni], false, false);
  }

  // epilogue: C layout row = j + 8*half, col = lane%16
#pragma unroll
  for (int mi = 0; mi < 2; ++mi)
#pragma unroll
    for (int ni = 0; ni < 2; ++ni)
#pragma unroll
      for (int j = 0; j < 8; ++j) {
        int m = bm + waveM * 32 + mi * 16 + j + 8 * half;
        int n = bn + waveN * 32 + ni * 16 + l16;
        float vv = acc[mi][ni][j];
        if (BIAS) vv += bias[n];
        if (RES)  vv += resid[(size_t)m * N + n];
        if (RELU) vv = fmaxf(vv, 0.0f);
        if (OUTBF) ((ushort_t*)outp)[(size_t)m * N + n] = f32_to_bf16(vv);
        else       ((float*)outp)[(size_t)m * N + n]    = vv;
      }
}

// ---------------------------------------------------------------------------
// Flash attention: one wave per (b, h, 16-row query tile). Keys in chunks of 32.
// q/k/v/o layout: [B*T, H*D] bf16, col = h*D + d. D=24 padded to K=32.
// All global loads are branch-free: rows are clamped (always in-bounds) and
// invalid data is zeroed with register selects (v_cndmask), never exec masks.
// ---------------------------------------------------------------------------
__global__ __launch_bounds__(32) void attn_kernel(
    const ushort_t* __restrict__ qb, const ushort_t* __restrict__ kbuf,
    const ushort_t* __restrict__ vb, ushort_t* __restrict__ ob) {
  __shared__ alignas(16) ushort_t Pl[16][40];   // probabilities tile [m][s'] (16x32 used)
  __shared__ alignas(16) ushort_t Vt[32][40];   // V transposed [d][s'] (24x32 used)

  const int blk  = blockIdx.x;
  const int qt   = blk & 63;        // T/16 = 64 query tiles
  const int bh   = blk >> 6;
  const int h    = bh & (H_ - 1);
  const int b    = bh >> 4;
  const int t0   = qt * 16;
  const int lane = threadIdx.x;
  const int half = lane >> 4;
  const int l16  = lane & 15;
  const float scale = 0.20412414523193154f;    // 1/sqrt(24)

  U2 z2; z2.a = 0u; z2.b = 0u;

  // Q fragment (A layout, K = padded D). Rows t0+l16 < T always.
  Frag qa;
  {
    const ushort_t* qrow = qb + ((size_t)(b * T_ + t0 + l16)) * C_ + h * D_;
    const ushort_t* p0 = qrow + half * 8;     // half0: d0..7 ; half1: d8..15
    U2 c0 = *(const U2*)(p0);
    U2 c1 = *(const U2*)(p0 + 4);
    const ushort_t* p1 = qrow + 16;           // d16..23 (always valid)
    U2 c2 = *(const U2*)(p1);
    U2 c3 = *(const U2*)(p1 + 4);
    qa.h2[0] = c0;
    qa.h2[1] = c1;
    qa.h2[2] = half ? z2 : c2;                // half1 upper K pad = 0
    qa.h2[3] = half ? z2 : c3;
  }

  float Mrun[8], Lrun[8];
#pragma unroll
  for (int j = 0; j < 8; ++j) { Mrun[j] = -1e30f; Lrun[j] = 0.0f; }
  v8f o0 = v8f_zero(), o1 = v8f_zero();

  for (int s0 = 0; s0 <= t0 + 15; s0 += 32) {
    // K^T fragments (B layout): col = s0(+16)+l16, K-half picks d {0..7,16..23} / {8..15,24..31}.
    Frag kf0, kf1;
#pragma unroll
    for (int tile = 0; tile < 2; ++tile) {
      int s  = s0 + tile * 16 + l16;
      bool ok = (s < T_);
      int sr = ok ? s : (T_ - 1);
      const ushort_t* krow = kbuf + ((size_t)(b * T_ + sr)) * C_ + h * D_;
      const ushort_t* p0 = krow + half * 8;           // d = 8*half + 0..7
      U2 c0 = *(const U2*)(p0);
      U2 c1 = *(const U2*)(p0 + 4);
      const ushort_t* p1 = krow + 16;                 // half0: d16..23; half1: dummy (zeroed)
      U2 c2 = *(const U2*)(p1);
      U2 c3 = *(const U2*)(p1 + 4);
      Frag& kf = tile ? kf1 : kf0;
      kf.h2[0] = ok ? c0 : z2;
      kf.h2[1] = ok ? c1 : z2;
      kf.h2[2] = (half || !ok) ? z2 : c2;
      kf.h2[3] = (half || !ok) ? z2 : c3;
    }
    v8f sc0 = __builtin_amdgcn_wmma_f32_16x16x32_bf16(false, qa.v, false, kf0.v, (short)0, v8f_zero(), false, false);
    v8f sc1 = __builtin_amdgcn_wmma_f32_16x16x32_bf16(false, qa.v, false, kf1.v, (short)0, v8f_zero(), false, false);

    // online softmax; row m = j + 8*half lives in the 16-lane half-group
    float p0v[8], p1v[8];
#pragma unroll
    for (int j = 0; j < 8; ++j) {
      int t  = t0 + j + 8 * half;
      int sA = s0 + l16;
      int sB = s0 + 16 + l16;
      float a = sc0[j] * scale;
      float c = sc1[j] * scale;
      bool mA = (sA > t), mB = (sB > t);
      if (mA) a = -1e30f;
      if (mB) c = -1e30f;
      float mx = fmaxf(a, c);
#pragma unroll
      for (int m = 8; m >= 1; m >>= 1) mx = fmaxf(mx, __shfl_xor(mx, m));
      float Mn = fmaxf(Mrun[j], mx);
      float e0 = mA ? 0.0f : __expf(a - Mn);
      float e1 = mB ? 0.0f : __expf(c - Mn);
      float rs = e0 + e1;
#pragma unroll
      for (int m = 8; m >= 1; m >>= 1) rs += __shfl_xor(rs, m);
      float cf = __expf(Mrun[j] - Mn);
      Lrun[j]  = Lrun[j] * cf + rs;
      Mrun[j]  = Mn;
      o0[j] *= cf;
      o1[j] *= cf;
      p0v[j] = e0;
      p1v[j] = e1;
    }

    __syncthreads();
    // P (C layout) -> LDS -> A-fragment layout transpose
#pragma unroll
    for (int j = 0; j < 8; ++j) {
      int m = j + 8 * half;
      Pl[m][l16]      = f32_to_bf16(p0v[j]);
      Pl[m][16 + l16] = f32_to_bf16(p1v[j]);
    }
    // V chunk (32 keys x 24 d) transposed into Vt[d][s']; branch-free loads
    {
      int s  = s0 + lane;
      bool ok = (s < T_);
      int sr = ok ? s : (T_ - 1);
      const ushort_t* vrow = vb + ((size_t)(b * T_ + sr)) * C_ + h * D_;
      U2 c[6];
#pragma unroll
      for (int i = 0; i < 6; ++i) c[i] = *(const U2*)(vrow + i * 4);
#pragma unroll
      for (int i = 0; i < 6; ++i) {
        U2 w = ok ? c[i] : z2;
        Vt[i * 4 + 0][lane] = (ushort_t)(w.a & 0xffffu);
        Vt[i * 4 + 1][lane] = (ushort_t)(w.a >> 16);
        Vt[i * 4 + 2][lane] = (ushort_t)(w.b & 0xffffu);
        Vt[i * 4 + 3][lane] = (ushort_t)(w.b >> 16);
      }
#pragma unroll
      for (int d = D_; d < 32; ++d) Vt[d][lane] = 0;
    }
    __syncthreads();

    Frag pa, vf0, vf1;
    pa.q[0] = *(const U4*)&Pl[l16][half * 16];
    pa.q[1] = *(const U4*)&Pl[l16][half * 16 + 8];
    int kf = half * 16;
    vf0.q[0] = *(const U4*)&Vt[l16][kf];
    vf0.q[1] = *(const U4*)&Vt[l16][kf + 8];
    vf1.q[0] = *(const U4*)&Vt[16 + l16][kf];
    vf1.q[1] = *(const U4*)&Vt[16 + l16][kf + 8];
    o0 = __builtin_amdgcn_wmma_f32_16x16x32_bf16(false, pa.v, false, vf0.v, (short)0, o0, false, false);
    o1 = __builtin_amdgcn_wmma_f32_16x16x32_bf16(false, pa.v, false, vf1.v, (short)0, o1, false, false);
    __syncthreads();
  }

  // normalize + store [b,t, h*D+d] bf16
#pragma unroll
  for (int j = 0; j < 8; ++j) {
    float inv = 1.0f / Lrun[j];
    int t = t0 + j + 8 * half;
    ushort_t* orow = ob + ((size_t)(b * T_ + t)) * C_ + h * D_;
    orow[l16] = f32_to_bf16(o0[j] * inv);
    int d2 = 16 + l16;
    if (d2 < D_) orow[d2] = f32_to_bf16(o1[j] * inv);
  }
}

// ---------------------------------------------------------------------------
extern "C" void kernel_launch(void* const* d_in, const int* in_sizes, int n_in,
                              void* d_out, int out_size, void* d_ws, size_t ws_size,
                              hipStream_t stream) {
  (void)in_sizes; (void)n_in; (void)out_size; (void)ws_size;

  const float* idx  = (const float*)d_in[0];
  const float* pos  = (const float*)d_in[1];
  const float* Wq   = (const float*)d_in[2];
  const float* Wk   = (const float*)d_in[3];
  const float* Wv   = (const float*)d_in[4];
  const float* Wo   = (const float*)d_in[5];
  const float* bo   = (const float*)d_in[6];
  const float* ln1g = (const float*)d_in[7];
  const float* ln1b = (const float*)d_in[8];
  const float* ln2g = (const float*)d_in[9];
  const float* ln2b = (const float*)d_in[10];
  const float* W1   = (const float*)d_in[11];
  const float* b1   = (const float*)d_in[12];
  const float* W2   = (const float*)d_in[13];
  const float* b2   = (const float*)d_in[14];
  const float* lnfg = (const float*)d_in[15];
  const float* lnfb = (const float*)d_in[16];

  char* ws = (char*)d_ws;
  size_t off = 0;
  auto alloc = [&](size_t bytes) -> void* {
    void* p = ws + off;
    off += (bytes + 255) & ~(size_t)255;
    return p;
  };

  float*    x    = (float*)   alloc((size_t)M_ * C_ * 4);
  ushort_t* hb   = (ushort_t*)alloc((size_t)M_ * C_ * 2);
  ushort_t* qbuf = (ushort_t*)alloc((size_t)M_ * C_ * 2);
  ushort_t* kbuf = (ushort_t*)alloc((size_t)M_ * C_ * 2);
  ushort_t* vbuf = (ushort_t*)alloc((size_t)M_ * C_ * 2);
  ushort_t* obuf = (ushort_t*)alloc((size_t)M_ * C_ * 2);
  ushort_t* ubuf = (ushort_t*)alloc((size_t)M_ * F_ * 2);
  ushort_t* wqp  = (ushort_t*)alloc((size_t)C_ * C_ * 2);
  ushort_t* wkp  = (ushort_t*)alloc((size_t)C_ * C_ * 2);
  ushort_t* wvp  = (ushort_t*)alloc((size_t)C_ * C_ * 2);
  ushort_t* wop  = (ushort_t*)alloc((size_t)C_ * C_ * 2);
  ushort_t* w1p  = (ushort_t*)alloc((size_t)C_ * F_ * 2);
  ushort_t* w2p  = (ushort_t*)alloc((size_t)F_ * C_ * 2);

  const dim3 blk256(256);
  const dim3 gridEmb((B_ * T_ * C_) / 256);
  const dim3 gridLN(M_ / 8);
  const dim3 gridPackQKV((C_ * H_ * D_ + 255) / 256);
  const dim3 gridPackC2((C_ * C_ + 255) / 256);
  const dim3 gridPackCF((C_ * F_ + 255) / 256);
  const dim3 gridGemmC(C_ / 64, M_ / 128);   // N=384
  const dim3 gridGemmF(F_ / 64, M_ / 128);   // N=1536
  const dim3 gridAttn(B_ * H_ * (T_ / 16));

  add_pos_kernel<<<gridEmb, blk256, 0, stream>>>(idx, pos, x);

  for (int l = 0; l < L_; ++l) {
    // ---- attention block ----
    ln_to_bf16_kernel<<<gridLN, blk256, 0, stream>>>(x, ln1g + l * C_, ln1b + l * C_, hb);

    pack_qkv_kernel<<<gridPackQKV, blk256, 0, stream>>>(Wq + (size_t)l * H_ * C_ * D_, wqp);
    pack_qkv_kernel<<<gridPackQKV, blk256, 0, stream>>>(Wk + (size_t)l * H_ * C_ * D_, wkp);
    pack_qkv_kernel<<<gridPackQKV, blk256, 0, stream>>>(Wv + (size_t)l * H_ * C_ * D_, wvp);

    gemm_bf16_kernel<false, false, false, true><<<gridGemmC, blk256, 0, stream>>>(
        hb, wqp, nullptr, nullptr, qbuf, M_, C_, C_);
    gemm_bf16_kernel<false, false, false, true><<<gridGemmC, blk256, 0, stream>>>(
        hb, wkp, nullptr, nullptr, kbuf, M_, C_, C_);
    gemm_bf16_kernel<false, false, false, true><<<gridGemmC, blk256, 0, stream>>>(
        hb, wvp, nullptr, nullptr, vbuf, M_, C_, C_);

    attn_kernel<<<gridAttn, dim3(32), 0, stream>>>(qbuf, kbuf, vbuf, obuf);

    pack_mat_kernel<<<gridPackC2, blk256, 0, stream>>>(Wo + (size_t)l * C_ * C_, wop, C_ * C_);
    // x = x + o @ Wo + bo
    gemm_bf16_kernel<true, true, false, false><<<gridGemmC, blk256, 0, stream>>>(
        obuf, wop, bo + l * C_, x, x, M_, C_, C_);

    // ---- MLP block ----
    ln_to_bf16_kernel<<<gridLN, blk256, 0, stream>>>(x, ln2g + l * C_, ln2b + l * C_, hb);

    pack_mat_kernel<<<gridPackCF, blk256, 0, stream>>>(W1 + (size_t)l * C_ * F_, w1p, C_ * F_);
    // u = relu(h2 @ W1 + b1)
    gemm_bf16_kernel<true, false, true, true><<<gridGemmF, blk256, 0, stream>>>(
        hb, w1p, b1 + l * F_, nullptr, ubuf, M_, F_, C_);

    pack_mat_kernel<<<gridPackCF, blk256, 0, stream>>>(W2 + (size_t)l * F_ * C_, w2p, F_ * C_);
    // x = x + u @ W2 + b2
    gemm_bf16_kernel<true, true, false, false><<<gridGemmC, blk256, 0, stream>>>(
        ubuf, w2p, b2 + l * C_, x, x, M_, C_, F_);
  }

  lnf_relu_kernel<<<gridLN, blk256, 0, stream>>>(x, lnfg, lnfb, (float*)d_out);
}
